// AKAZESparseBADSinkhornMatcher_30374008717909
// MI455X (gfx1250) — matrix operand order repaired and checked by reading, not compiled, not run on hardware
//
#include <hip/hip_runtime.h>
#include <hip/hip_bf16.h>

// ---------------- constants (match reference) ----------------
constexpr int   cB = 2;
constexpr int   cH = 960;
constexpr int   cW = 1280;
constexpr int   cN = cB * cH * cW;
constexpr int   cK = 1024;
constexpr int   cP = 256;
constexpr int   cZ = 1025;              // K+1
constexpr float cKAPPA   = 0.05f;
constexpr float cTAU     = 0.2f;
constexpr float cDET_THR = 0.001f;
constexpr float cORI_SIG = 2.5f;
constexpr float cNORM    = -7.6246189861593985f;  // -log(2*K)
constexpr float cLOGK    =  6.9314718055994531f;  // log(K)

typedef __attribute__((ext_vector_type(16))) _Float16 v16h;
typedef __attribute__((ext_vector_type(8)))  float    v8f;

// ---------------- small helpers ----------------
__device__ __forceinline__ float ldz(const float* p, int b, int y, int x) {
  if (y < 0 || y >= cH || x < 0 || x >= cW) return 0.0f;
  return p[((size_t)b * cH + y) * cW + x];
}
__device__ __forceinline__ float lde(const float* p, int b, int y, int x) {
  y = y < 0 ? 0 : (y >= cH ? cH - 1 : y);
  x = x < 0 ? 0 : (x >= cW ? cW - 1 : x);
  return p[((size_t)b * cH + y) * cW + x];
}
__device__ __forceinline__ float sobelX(const float* p, int b, int y, int x) {
  return (-ldz(p,b,y-1,x-1) +      ldz(p,b,y-1,x+1)
          -2.0f*ldz(p,b,y,x-1) + 2.0f*ldz(p,b,y,x+1)
          -ldz(p,b,y+1,x-1) +      ldz(p,b,y+1,x+1)) * 0.125f;
}
__device__ __forceinline__ float sobelY(const float* p, int b, int y, int x) {
  return (-ldz(p,b,y-1,x-1) - 2.0f*ldz(p,b,y-1,x) - ldz(p,b,y-1,x+1)
          +ldz(p,b,y+1,x-1) + 2.0f*ldz(p,b,y+1,x) + ldz(p,b,y+1,x+1)) * 0.125f;
}

// ---------------- utility fills / copies ----------------
__global__ void k_copy(float* dst, const float* src, int n) {
  int i = blockIdx.x * blockDim.x + threadIdx.x;
  if (i < n) dst[i] = src[i];
}
__global__ void k_fill_f32(float* p, float v, int n) {
  int i = blockIdx.x * blockDim.x + threadIdx.x;
  if (i < n) p[i] = v;
}
__global__ void k_fill_i32(int* p, int v, int n) {
  int i = blockIdx.x * blockDim.x + threadIdx.x;
  if (i < n) p[i] = v;
}

// ---------------- AKAZE: diffusion ----------------
__global__ void k_diffusivity(const float* __restrict__ L, float* __restrict__ g) {
  int i = blockIdx.x * blockDim.x + threadIdx.x;
  if (i >= cN) return;
  int b = i / (cH * cW), r = i % (cH * cW), y = r / cW, x = r % cW;
  float lx = sobelX(L, b, y, x), ly = sobelY(L, b, y, x);
  g[i] = 1.0f / (1.0f + (lx * lx + ly * ly) / (cKAPPA * cKAPPA));
}

__global__ void k_fed(const float* __restrict__ L, const float* __restrict__ g,
                      float* __restrict__ out) {
  int i = blockIdx.x * blockDim.x + threadIdx.x;
  if (i >= cN) return;
  int b = i / (cH * cW), r = i % (cH * cW), y = r / cW, x = r % cW;
  float L0 = L[i], g0 = g[i];
  float Ln = lde(L,b,y-1,x), Ls = lde(L,b,y+1,x), Lw = lde(L,b,y,x-1), Le = lde(L,b,y,x+1);
  float gn = lde(g,b,y-1,x), gs = lde(g,b,y+1,x), gw = lde(g,b,y,x-1), ge = lde(g,b,y,x+1);
  float flux = 0.5f * ((g0+gn)*(Ln-L0) + (g0+gs)*(Ls-L0) + (g0+gw)*(Lw-L0) + (g0+ge)*(Le-L0));
  out[i] = L0 + cTAU * flux;
}

__global__ void k_sobelx_map(const float* __restrict__ L, float* __restrict__ out) {
  int i = blockIdx.x * blockDim.x + threadIdx.x;
  if (i >= cN) return;
  int b = i / (cH * cW), r = i % (cH * cW), y = r / cW, x = r % cW;
  out[i] = sobelX(L, b, y, x);
}

__global__ void k_hessian(const float* __restrict__ L, const float* __restrict__ Lx,
                          float* __restrict__ resp) {
  int i = blockIdx.x * blockDim.x + threadIdx.x;
  if (i >= cN) return;
  int b = i / (cH * cW), r = i % (cH * cW), y = r / cW, x = r % cW;
  float c0  = L[i];
  float lxx = ldz(L,b,y,x-1) - 2.0f*c0 + ldz(L,b,y,x+1);
  float lyy = ldz(L,b,y-1,x) - 2.0f*c0 + ldz(L,b,y+1,x);
  float lxy = sobelY(Lx, b, y, x);
  float det = lxx * lyy - lxy * lxy;
  resp[i] = fmaxf(resp[i], det);
}

// ---------------- threshold + NMS ----------------
__global__ void k_thr_nms5(const float* __restrict__ R, float* __restrict__ out) {
  int i = blockIdx.x * blockDim.x + threadIdx.x;
  if (i >= cN) return;
  int b = i / (cH * cW), r = i % (cH * cW), y = r / cW, x = r % cW;
  float c0 = R[i];
  float t  = c0 > cDET_THR ? c0 : 0.0f;
  float m  = -3.4e38f;
  for (int dy = -2; dy <= 2; ++dy)
    for (int dx = -2; dx <= 2; ++dx) {
      int yy = y + dy, xx = x + dx;
      if (yy < 0 || yy >= cH || xx < 0 || xx >= cW) continue;
      float v = R[((size_t)b * cH + yy) * cW + xx];
      v = v > cDET_THR ? v : 0.0f;
      m = fmaxf(m, v);
    }
  out[i] = (t >= m - 1e-7f) ? t : 0.0f;
}

__global__ void k_nms7(const float* __restrict__ R, float* __restrict__ out) {
  int i = blockIdx.x * blockDim.x + threadIdx.x;
  if (i >= cN) return;
  int b = i / (cH * cW), r = i % (cH * cW), y = r / cW, x = r % cW;
  float c0 = R[i];
  float m  = -3.4e38f;
  for (int dy = -3; dy <= 3; ++dy)
    for (int dx = -3; dx <= 3; ++dx) {
      int yy = y + dy, xx = x + dx;
      if (yy < 0 || yy >= cH || xx < 0 || xx >= cW) continue;
      m = fmaxf(m, R[((size_t)b * cH + yy) * cW + xx]);
    }
  float s = (c0 >= m - 1e-7f) ? c0 : 0.0f;
  out[i] = s > 0.0f ? s : 0.0f;
}

// ---------------- top-K selection (exponent-bucket radix select) ----------------
__global__ void k_hist(const float* __restrict__ S, int* __restrict__ hist) {
  int i = blockIdx.x * blockDim.x + threadIdx.x;
  if (i >= cN) return;
  float s = S[i];
  if (s > 0.0f) {
    int b = i / (cH * cW);
    unsigned bits = __float_as_uint(s);
    int bin = (int)((bits >> 23) & 0xFFu);
    atomicAdd(&hist[b * 256 + bin], 1);
  }
}

__global__ void k_pick(const int* __restrict__ hist, int* __restrict__ sel) {
  int b = threadIdx.x;
  if (b >= cB) return;
  int cum = 0, cut = -1, rem = 0;
  for (int bin = 255; bin >= 0; --bin) {
    int h = hist[b * 256 + bin];
    if (cum + h >= cK) { cut = bin; rem = cK - cum; break; }
    cum += h;
  }
  sel[b * 4 + 0] = cut;
  sel[b * 4 + 1] = rem;
  sel[b * 4 + 2] = 0;   // slot counter
  sel[b * 4 + 3] = 0;   // cut-bucket counter
}

__global__ void k_collect(const float* __restrict__ S, int* __restrict__ sel,
                          float* __restrict__ candS, int* __restrict__ candI) {
  int i = blockIdx.x * blockDim.x + threadIdx.x;
  if (i >= cN) return;
  float s = S[i];
  if (s <= 0.0f) return;
  int b = i / (cH * cW);
  int local = i % (cH * cW);
  unsigned bits = __float_as_uint(s);
  int bin = (int)((bits >> 23) & 0xFFu);
  int* sb = sel + b * 4;
  bool take = false;
  if (bin > sb[0]) take = true;
  else if (bin == sb[0]) { if (atomicAdd(&sb[3], 1) < sb[1]) take = true; }
  if (take) {
    int slot = atomicAdd(&sb[2], 1);
    if (slot < cK) { candS[b * cK + slot] = s; candI[b * cK + slot] = local; }
  }
}

__global__ __launch_bounds__(1024) void k_sort_select(const float* __restrict__ candS,
                                                      const int* __restrict__ candI,
                                                      float* __restrict__ kout,
                                                      float* __restrict__ kscore) {
  __shared__ float ss[1024];
  __shared__ int   si[1024];
  int b = blockIdx.x, t = threadIdx.x;
  ss[t] = candS[b * cK + t];
  si[t] = candI[b * cK + t];
  __syncthreads();
  for (int k = 2; k <= 1024; k <<= 1) {
    for (int j = k >> 1; j > 0; j >>= 1) {
      int ixj = t ^ j;
      if (ixj > t) {
        bool desc = ((t & k) == 0);          // descending segments -> overall descending
        float a = ss[t], c = ss[ixj];
        bool sw = desc ? (a < c) : (a > c);
        if (sw) {
          ss[t] = c; ss[ixj] = a;
          int tmp = si[t]; si[t] = si[ixj]; si[ixj] = tmp;
        }
      }
      __syncthreads();
    }
  }
  float s = ss[t];
  int   id = si[t];
  bool  valid = (s > 0.0f) && (id >= 0);
  kout[((size_t)b * cK + t) * 2 + 0] = valid ? (float)(id / cW) : -1.0f;
  kout[((size_t)b * cK + t) * 2 + 1] = valid ? (float)(id % cW) : -1.0f;
  kscore[b * cK + t] = valid ? s : 0.0f;
}

// ---------------- gradients for orientation ----------------
__global__ void k_grad(const float* __restrict__ img, float* __restrict__ gx,
                       float* __restrict__ gy) {
  int i = blockIdx.x * blockDim.x + threadIdx.x;
  if (i >= cN) return;
  int b = i / (cH * cW), r = i % (cH * cW), y = r / cW, x = r % cW;
  gx[i] = sobelX(img, b, y, x);
  gy[i] = sobelY(img, b, y, x);
}

// ---------------- BAD descriptor ----------------
__device__ __forceinline__ float boxsamp(const float* img, int b, float px, float py, int r) {
  int xi = (int)rintf(fminf(fmaxf(px, 0.0f), (float)(cW - 1)));
  int yi = (int)rintf(fminf(fmaxf(py, 0.0f), (float)(cH - 1)));
  float s = 0.0f;
  for (int dy = -r; dy <= r; ++dy)
    for (int dx = -r; dx <= r; ++dx)
      s += ldz(img, b, yi + dy, xi + dx);
  float w = (float)(2 * r + 1);
  return s / (w * w);
}

__global__ __launch_bounds__(256) void k_descriptor(
    const float* __restrict__ img, const float* __restrict__ gx, const float* __restrict__ gy,
    const float* __restrict__ kpts, const float* __restrict__ kscore,
    const float* __restrict__ offs, const float* __restrict__ thr, const int* __restrict__ radii,
    _Float16* __restrict__ desc, float* __restrict__ sq) {
  int blk = blockIdx.x;
  int b = blk / cK, k = blk % cK, t = threadIdx.x;
  __shared__ float rA[256], rB[256];
  __shared__ float sh[2];

  float ky = kpts[((size_t)b * cK + k) * 2 + 0];
  float kx = kpts[((size_t)b * cK + k) * 2 + 1];
  float valid = kscore[b * cK + k] > 0.0f ? 1.0f : 0.0f;
  float yc = fminf(fmaxf(ky, 0.0f), (float)(cH - 1));
  float xc = fminf(fmaxf(kx, 0.0f), (float)(cW - 1));
  int yi = (int)rintf(yc), xi = (int)rintf(xc);

  // orientation: Gaussian-weighted Sobel response (normalization cancels in atan2)
  float ax = 0.0f, ay = 0.0f;
  if (t < 225) {
    int dy = t / 15 - 7, dx = t % 15 - 7;
    float w = expf(-(float)(dx * dx + dy * dy) / (2.0f * cORI_SIG * cORI_SIG));
    ax = w * ldz(gx, b, yi + dy, xi + dx);
    ay = w * ldz(gy, b, yi + dy, xi + dx);
  }
  rA[t] = ax; rB[t] = ay;
  __syncthreads();
  for (int s2 = 128; s2 > 0; s2 >>= 1) {
    if (t < s2) { rA[t] += rA[t + s2]; rB[t] += rB[t + s2]; }
    __syncthreads();
  }
  if (t == 0) {
    float th = atan2f(rB[0], rA[0]);
    sh[0] = cosf(th); sh[1] = sinf(th);
  }
  __syncthreads();
  float cth = sh[0], sth = sh[1];

  float ox1 = offs[t * 4 + 0], oy1 = offs[t * 4 + 1];
  float ox2 = offs[t * 4 + 2], oy2 = offs[t * 4 + 3];
  float px1 = xc + cth * ox1 - sth * oy1;
  float py1 = yc + sth * ox1 + cth * oy1;
  float px2 = xc + cth * ox2 - sth * oy2;
  float py2 = yc + sth * ox2 + cth * oy2;
  int r = radii[t]; r = r < 0 ? 0 : (r > 5 ? 5 : r);
  float v = (boxsamp(img, b, px1, py1, r) - boxsamp(img, b, px2, py2, r) - thr[t]) * valid;

  __syncthreads();
  rA[t] = v * v;
  __syncthreads();
  for (int s2 = 128; s2 > 0; s2 >>= 1) {
    if (t < s2) rA[t] += rA[t + s2];
    __syncthreads();
  }
  float ssum = rA[0];
  float n = sqrtf(ssum) + 1e-8f;
  float d = v / n;
  desc[((size_t)b * cK + k) * cP + t] = (_Float16)d;
  if (t == 0) sq[b * cK + k] = ssum / (n * n);
}

// ---------------- Sinkhorn: Z init + WMMA distance GEMM ----------------
__global__ void k_zborder(float* __restrict__ Z) {
  int i = blockIdx.x * blockDim.x + threadIdx.x;
  if (i >= cB * cZ) return;
  int b = i / cZ, j = i % cZ;
  size_t base = (size_t)b * cZ * cZ;
  Z[base + (size_t)cK * cZ + j] = 1.0f;   // UNUSED_SCORE / EPSILON
  Z[base + (size_t)j * cZ + cK] = 1.0f;
}

// One 16x16 C tile per wave; K=256 in 8 steps of v_wmma_f32_16x16x32_f16.
// A lane layout (ISA 7.12.2, 16-bit A 16x32): lanes0-15 K{0..7,16..23}, lanes16-31 K{8..15,24..31}
// B lane layout (16-bit B, KxN): lanes0-15 K{0..7,16..23}, lanes16-31 K{8..15,24..31}, N = lane&15
// C layout: VGPR r -> M = r (+8 for lanes16-31), N = lane&15
__global__ __launch_bounds__(256) void k_gemm_dist(
    const _Float16* __restrict__ dA, const _Float16* __restrict__ dB,
    const float* __restrict__ sqA, const float* __restrict__ sqB,
    float* __restrict__ Z) {
  int wave = (blockIdx.x * blockDim.x + threadIdx.x) >> 5;
  int lane = threadIdx.x & 31;
  int b  = wave / (64 * 64);
  int tt = wave % (64 * 64);
  int i0 = (tt / 64) * 16;
  int j0 = (tt % 64) * 16;
  const _Float16* Abase = dA + (size_t)b * cK * cP;
  const _Float16* Bbase = dB + (size_t)b * cK * cP;
  int rowA = i0 + (lane & 15);
  int colB = j0 + (lane & 15);
  int hiA  = (lane >> 4) * 8;
  int hiB  = (lane >> 4) * 8;

  v8f acc = {};
  for (int kk = 0; kk < cP; kk += 32) {
    const _Float16* ap = Abase + (size_t)rowA * cP + kk + hiA;
    const _Float16* bp = Bbase + (size_t)colB * cP + kk + hiB;
    v16h a, bf;
#pragma unroll
    for (int j = 0; j < 4; ++j) { a[2*j] = ap[2*j]; a[2*j+1] = ap[2*j+1]; }
#pragma unroll
    for (int j = 4; j < 8; ++j) { a[2*j] = ap[16 + 2*(j-4)]; a[2*j+1] = ap[16 + 2*(j-4) + 1]; }
#pragma unroll
    for (int j = 0; j < 8; ++j) { bf[2*j] = bp[(j >> 2) * 16 + 2*(j & 3)]; bf[2*j+1] = bp[(j >> 2) * 16 + 2*(j & 3) + 1]; }
    acc = __builtin_amdgcn_wmma_f32_16x16x32_f16(false, a, false, bf,
                                                 (short)0, acc, false, false);
  }
#pragma unroll
  for (int r = 0; r < 8; ++r) {
    int i = i0 + r + ((lane >> 4) << 3);
    int j = j0 + (lane & 15);
    float dot = acc[r];
    float d2  = sqA[b * cK + i] + sqB[b * cK + j] - 2.0f * dot;
    float dist = sqrtf(fmaxf(d2, 0.0f) + 1e-12f);
    Z[(size_t)b * cZ * cZ + (size_t)i * cZ + j] = -dist;  // EPSILON = 1
  }
}

// ---------------- Sinkhorn iterations ----------------
__global__ __launch_bounds__(256) void k_sink_row(const float* __restrict__ Z,
                                                  const float* __restrict__ vv,
                                                  float* __restrict__ uu) {
  int b = blockIdx.x / cZ, i = blockIdx.x % cZ, t = threadIdx.x;
  const float* zr = Z + (size_t)b * cZ * cZ + (size_t)i * cZ;
  const float* vb = vv + b * cZ;
  __shared__ float sm[256];
  float m = -3.4e38f;
  for (int j = t; j < cZ; j += 256) m = fmaxf(m, zr[j] + vb[j]);
  sm[t] = m; __syncthreads();
  for (int s = 128; s > 0; s >>= 1) { if (t < s) sm[t] = fmaxf(sm[t], sm[t + s]); __syncthreads(); }
  m = sm[0]; __syncthreads();
  float s = 0.0f;
  for (int j = t; j < cZ; j += 256) s += expf(zr[j] + vb[j] - m);
  sm[t] = s; __syncthreads();
  for (int s2 = 128; s2 > 0; s2 >>= 1) { if (t < s2) sm[t] += sm[t + s2]; __syncthreads(); }
  if (t == 0) {
    float lse = m + logf(sm[0]);
    float lmu = cNORM + ((i == cK) ? cLOGK : 0.0f);
    uu[b * cZ + i] = lmu - lse;
  }
}

__global__ __launch_bounds__(256) void k_sink_col(const float* __restrict__ Z,
                                                  const float* __restrict__ uu,
                                                  float* __restrict__ vv) {
  int b = blockIdx.x / cZ, j = blockIdx.x % cZ, t = threadIdx.x;
  const float* zc = Z + (size_t)b * cZ * cZ + j;
  const float* ub = uu + b * cZ;
  __shared__ float sm[256];
  float m = -3.4e38f;
  for (int i = t; i < cZ; i += 256) m = fmaxf(m, zc[(size_t)i * cZ] + ub[i]);
  sm[t] = m; __syncthreads();
  for (int s = 128; s > 0; s >>= 1) { if (t < s) sm[t] = fmaxf(sm[t], sm[t + s]); __syncthreads(); }
  m = sm[0]; __syncthreads();
  float s = 0.0f;
  for (int i = t; i < cZ; i += 256) s += expf(zc[(size_t)i * cZ] + ub[i] - m);
  sm[t] = s; __syncthreads();
  for (int s2 = 128; s2 > 0; s2 >>= 1) { if (t < s2) sm[t] += sm[t + s2]; __syncthreads(); }
  if (t == 0) {
    float lse = m + logf(sm[0]);
    float lnu = cNORM + ((j == cK) ? cLOGK : 0.0f);
    vv[b * cZ + j] = lnu - lse;
  }
}

__global__ void k_probs(float* __restrict__ Z, const float* __restrict__ uu,
                        const float* __restrict__ vv, int n) {
  int idx = blockIdx.x * blockDim.x + threadIdx.x;
  if (idx >= n) return;
  int b = idx / (cZ * cZ);
  int r = idx % (cZ * cZ);
  int i = r / cZ, j = r % cZ;
  Z[idx] = expf(Z[idx] + uu[b * cZ + i] + vv[b * cZ + j] - cNORM);
}

// ---------------- host orchestration ----------------
extern "C" void kernel_launch(void* const* d_in, const int* in_sizes, int n_in,
                              void* d_out, int out_size, void* d_ws, size_t ws_size,
                              hipStream_t stream) {
  (void)in_sizes; (void)n_in; (void)out_size; (void)ws_size;
  const float* img1 = (const float*)d_in[0];
  const float* img2 = (const float*)d_in[1];
  const float* offs = (const float*)d_in[2];
  const float* thr  = (const float*)d_in[3];
  const int*   rad  = (const int*)d_in[4];

  float* out   = (float*)d_out;
  float* k1out = out;                      // (B,K,2)
  float* k2out = out + (size_t)cB * cK * 2;
  float* Zb    = out + (size_t)2 * cB * cK * 2;  // probs region, (B, 1025, 1025)

  char*  w  = (char*)d_ws;
  size_t NB = (size_t)cN * sizeof(float);
  auto align256 = [](size_t x) { return (x + 255) & ~(size_t)255; };
  float* bufA = (float*)(w + 0 * NB);
  float* bufB = (float*)(w + 1 * NB);
  float* bufG = (float*)(w + 2 * NB);
  float* bufR = (float*)(w + 3 * NB);
  float* bufS = (float*)(w + 4 * NB);
  size_t off = align256(5 * NB);
  _Float16* desc1 = (_Float16*)(w + off); off = align256(off + (size_t)cB * cK * cP * 2);
  _Float16* desc2 = (_Float16*)(w + off); off = align256(off + (size_t)cB * cK * cP * 2);
  float* sq1     = (float*)(w + off); off = align256(off + (size_t)cB * cK * 4);
  float* sq2     = (float*)(w + off); off = align256(off + (size_t)cB * cK * 4);
  float* kscore1 = (float*)(w + off); off = align256(off + (size_t)cB * cK * 4);
  float* kscore2 = (float*)(w + off); off = align256(off + (size_t)cB * cK * 4);
  float* uu      = (float*)(w + off); off = align256(off + (size_t)cB * cZ * 4);
  float* vv      = (float*)(w + off); off = align256(off + (size_t)cB * cZ * 4);
  int*   hist    = (int*)(w + off);   off = align256(off + (size_t)cB * 256 * 4);
  int*   sel     = (int*)(w + off);   off = align256(off + 256);
  float* candS   = (float*)(w + off); off = align256(off + (size_t)cB * cK * 4);
  int*   candI   = (int*)(w + off);   off = align256(off + (size_t)cB * cK * 4);

  const int gbN = (cN + 255) / 256;

  auto detect = [&](const float* img, float* kout, float* kscore,
                    _Float16* desc, float* sq) {
    k_copy<<<gbN, 256, 0, stream>>>(bufA, img, cN);
    k_fill_f32<<<gbN, 256, 0, stream>>>(bufR, 0.0f, cN);
    float* Lc = bufA;
    float* Ln = bufB;
    for (int sc = 0; sc < 3; ++sc) {
      for (int it = 0; it < 3; ++it) {
        k_diffusivity<<<gbN, 256, 0, stream>>>(Lc, bufG);
        k_fed<<<gbN, 256, 0, stream>>>(Lc, bufG, Ln);
        float* tmp = Lc; Lc = Ln; Ln = tmp;
      }
      k_sobelx_map<<<gbN, 256, 0, stream>>>(Lc, bufG);
      k_hessian<<<gbN, 256, 0, stream>>>(Lc, bufG, bufR);
    }
    k_thr_nms5<<<gbN, 256, 0, stream>>>(bufR, bufS);
    k_nms7<<<gbN, 256, 0, stream>>>(bufS, bufR);

    k_fill_i32<<<(cB * 256 + 255) / 256, 256, 0, stream>>>(hist, 0, cB * 256);
    k_fill_f32<<<(cB * cK + 255) / 256, 256, 0, stream>>>(candS, 0.0f, cB * cK);
    k_fill_i32<<<(cB * cK + 255) / 256, 256, 0, stream>>>(candI, -1, cB * cK);
    k_hist<<<gbN, 256, 0, stream>>>(bufR, hist);
    k_pick<<<1, 32, 0, stream>>>(hist, sel);
    k_collect<<<gbN, 256, 0, stream>>>(bufR, sel, candS, candI);
    k_sort_select<<<cB, 1024, 0, stream>>>(candS, candI, kout, kscore);

    k_grad<<<gbN, 256, 0, stream>>>(img, bufG, bufA);  // gx->G, gy->A (L no longer needed)
    k_descriptor<<<cB * cK, 256, 0, stream>>>(img, bufG, bufA, kout, kscore,
                                              offs, thr, rad, desc, sq);
  };

  detect(img1, k1out, kscore1, desc1, sq1);
  detect(img2, k2out, kscore2, desc2, sq2);

  k_zborder<<<(cB * cZ + 255) / 256, 256, 0, stream>>>(Zb);
  // B*64*64 = 8192 tiles, one wave each, 8 waves per 256-thread block -> 1024 blocks
  k_gemm_dist<<<(cB * 64 * 64) / 8, 256, 0, stream>>>(desc1, desc2, sq1, sq2, Zb);

  k_fill_f32<<<(cB * cZ + 255) / 256, 256, 0, stream>>>(uu, 0.0f, cB * cZ);
  k_fill_f32<<<(cB * cZ + 255) / 256, 256, 0, stream>>>(vv, 0.0f, cB * cZ);
  for (int it = 0; it < 20; ++it) {
    k_sink_row<<<cB * cZ, 256, 0, stream>>>(Zb, vv, uu);
    k_sink_col<<<cB * cZ, 256, 0, stream>>>(Zb, uu, vv);
  }
  int np = cB * cZ * cZ;
  k_probs<<<(np + 255) / 256, 256, 0, stream>>>(Zb, uu, vv, np);
}
